// Mamba_14611478741649
// MI455X (gfx1250) — matrix operand order repaired and checked by reading, not compiled, not run on hardware
//
#include <hip/hip_runtime.h>
#include <cstddef>
#include <cstdint>

#ifndef __has_builtin
#define __has_builtin(x) 0
#endif

// ---- CDNA5 WMMA vector types ----
typedef __attribute__((ext_vector_type(16))) __bf16 v16bf;
typedef __attribute__((ext_vector_type(8)))  __bf16 v8bf;
typedef __attribute__((ext_vector_type(8)))  float  v8f;
typedef __attribute__((ext_vector_type(4)))  int    v4i;

#define D_MODEL 1024
#define D_INNER 2048
#define D_STATE 16
#define D_CONV  4
#define DT_RANK 64
#define SEQLEN  2048

// LDS B-tile: 64 rows (n) x 32 k bf16, row padded to 40 elems (80B) for bank spread.
#define BROW 40

#if __has_builtin(__builtin_amdgcn_global_load_async_to_lds_b128)
#define ASYNC_STAGE 1
typedef __attribute__((address_space(1))) v4i* gbl_b128;
typedef __attribute__((address_space(3))) v4i* lds_b128;
#endif

__device__ __forceinline__ float softplus_f(float x) {
  return (x > 20.0f) ? x : log1pf(expf(x));
}
__device__ __forceinline__ float silu_f(float x) {
  return x / (1.0f + expf(-x));
}
__device__ __forceinline__ void wait_async_lds() {
#if defined(ASYNC_STAGE)
#if __has_builtin(__builtin_amdgcn_s_wait_asynccnt)
  __builtin_amdgcn_s_wait_asynccnt(0);
#else
  asm volatile("s_wait_asynccnt 0x0" ::: "memory");
#endif
#endif
}

// ---------------- f32 -> bf16 bulk cast (8 elems / thread) ---------------------------
__global__ __launch_bounds__(256)
void cast_f32_to_bf16(const float* __restrict__ src, __bf16* __restrict__ dst) {
  const size_t i = ((size_t)blockIdx.x * 256 + threadIdx.x) * 8;
  const float4 a = *(const float4*)(src + i);
  const float4 b = *(const float4*)(src + i + 4);
  v8bf o;
  o[0] = (__bf16)a.x; o[1] = (__bf16)a.y; o[2] = (__bf16)a.z; o[3] = (__bf16)a.w;
  o[4] = (__bf16)b.x; o[5] = (__bf16)b.y; o[6] = (__bf16)b.z; o[7] = (__bf16)b.w;
  *(v8bf*)(dst + i) = o;
}

// ---------------- LDS-staged bf16 WMMA GEMM (block tile 128x64) ----------------------
// 8 waves share one n-range; B tile (64x32 bf16) staged in LDS once per k-step,
// double-buffered, GLOBAL_LOAD_ASYNC_TO_LDS_B128 when available (ASYNCcnt-tracked).
// Fragment layouts per CDNA5 ISA 7.12.2 (wave32).
template<int ACT>
__global__ __launch_bounds__(256)
void gemm_bf16_wmma_lds(const __bf16* __restrict__ A, const __bf16* __restrict__ W,
                        float* __restrict__ C, __bf16* __restrict__ Cb,
                        int M, int N, int K, int lda, int ldb, int ldc,
                        const float* __restrict__ bias) {
  __shared__ __align__(16) __bf16 btile[2][64 * BROW];

  const int lane  = threadIdx.x & 31;
  const int waveI = threadIdx.x >> 5;               // 0..7
  const int nBlk  = N / 64;
  const int bm    = blockIdx.x / nBlk;
  const int bn    = blockIdx.x - bm * nBlk;
  const int m0    = bm * 128 + waveI * 16;
  const int n0    = bn * 64;

  const int arow = m0 + (lane & 15);
  const int akb  = (lane < 16) ? 0 : 8;
  const int bcol = lane & 15;
  const int bkb  = (lane < 16) ? 0 : 8;              // element offset into 32-K row

  const __bf16* aBase = A + (size_t)arow * lda + akb;

  // staging decomposition: 256 threads, one b128 each: row n = tid/4, k-seg = tid%4
  const int sn = threadIdx.x >> 2;
  const int sk = threadIdx.x & 3;
  const __bf16* sBase = W + (size_t)(n0 + sn) * ldb + sk * 8;
  __bf16* sDst0 = &btile[0][sn * BROW + sk * 8];
  __bf16* sDst1 = &btile[1][sn * BROW + sk * 8];

  auto stageB = [&](int buf, int k0) {
    const __bf16* src = sBase + k0;
    __bf16* dst = buf ? sDst1 : sDst0;
#if defined(ASYNC_STAGE)
    __builtin_amdgcn_global_load_async_to_lds_b128(
        (gbl_b128)(uintptr_t)src,
        (lds_b128)(uint32_t)(uintptr_t)dst, 0, 0);
#else
    *(v8bf*)dst = *(const v8bf*)src;
#endif
  };

  v8f acc[4];
#pragma unroll
  for (int j = 0; j < 4; ++j)
#pragma unroll
    for (int r = 0; r < 8; ++r) acc[j][r] = 0.0f;

  stageB(0, 0);
  wait_async_lds();
  __syncthreads();

  for (int k0 = 0; k0 < K; k0 += 32) {
    const int cur  = (k0 >> 5) & 1;
    const bool more = (k0 + 32) < K;
    if (more) stageB(cur ^ 1, k0 + 32);

    // A fragment: private per wave, straight from global (L2-resident)
    const v8bf a_lo = *(const v8bf*)(aBase + k0);
    const v8bf a_hi = *(const v8bf*)(aBase + k0 + 16);
    if (more) __builtin_prefetch(aBase + k0 + 32, 0, 1);   // global_prefetch_b8
    const v16bf af = __builtin_shufflevector(a_lo, a_hi,
        0,1,2,3,4,5,6,7,8,9,10,11,12,13,14,15);

#pragma unroll
    for (int j = 0; j < 4; ++j) {
      const __bf16* bp = &btile[cur][(j * 16 + bcol) * BROW + bkb];
      const v8bf b_lo = *(const v8bf*)(bp);      // ds_load_b128
      const v8bf b_hi = *(const v8bf*)(bp + 16); // ds_load_b128
      const v16bf bfr = __builtin_shufflevector(b_lo, b_hi,
          0,1,2,3,4,5,6,7,8,9,10,11,12,13,14,15);
      acc[j] = __builtin_amdgcn_wmma_f32_16x16x32_bf16(
          false, af, false, bfr, (short)0, acc[j], false, false);
    }

    if (more) {
      wait_async_lds();
      __syncthreads();
    }
  }

  const int crow0 = m0 + ((lane < 16) ? 0 : 8);
#pragma unroll
  for (int j = 0; j < 4; ++j) {
    const int n = n0 + j * 16 + bcol;
    const float bmod = (ACT == 1) ? bias[n] : 0.0f;
#pragma unroll
    for (int r = 0; r < 8; ++r) {
      float v = acc[j][r];
      if (ACT == 1) v = softplus_f(v + bmod);
      C[(size_t)(crow0 + r) * ldc + n] = v;
      if (Cb) Cb[(size_t)(crow0 + r) * ldc + n] = (__bf16)v;
    }
  }
}

// ---------------- per-wave bf16 WMMA GEMM (for small N, e.g. 96) ---------------------
template<int NT, int ACT>
__global__ __launch_bounds__(256)
void gemm_bf16_wmma(const __bf16* __restrict__ A, const __bf16* __restrict__ W,
                    float* __restrict__ C, __bf16* __restrict__ Cb,
                    int M, int N, int K, int lda, int ldb, int ldc,
                    const float* __restrict__ bias) {
  const int lane = threadIdx.x & 31;
  const int wave = blockIdx.x * (blockDim.x >> 5) + (threadIdx.x >> 5);
  const int nSup = N / (16 * NT);
  const int tm   = wave / nSup;
  const int tn   = wave - tm * nSup;
  const int m0   = tm * 16;
  const int n0   = tn * 16 * NT;
  if (m0 >= M) return;

  const int arow = m0 + (lane & 15);
  const int akb  = (lane < 16) ? 0 : 8;
  const int bcol = lane & 15;
  const int bkb  = (lane < 16) ? 0 : 8;
  const __bf16* aBase = A + (size_t)arow * lda + akb;

  v8f acc[NT];
#pragma unroll
  for (int j = 0; j < NT; ++j)
#pragma unroll
    for (int r = 0; r < 8; ++r) acc[j][r] = 0.0f;

#pragma unroll 2
  for (int k0 = 0; k0 < K; k0 += 32) {
    const v8bf a_lo = *(const v8bf*)(aBase + k0);
    const v8bf a_hi = *(const v8bf*)(aBase + k0 + 16);
    v8bf blo[NT], bhi[NT];
#pragma unroll
    for (int j = 0; j < NT; ++j) {
      const __bf16* wp = W + (size_t)(n0 + j * 16 + bcol) * ldb + k0 + bkb;
      blo[j] = *(const v8bf*)(wp);
      bhi[j] = *(const v8bf*)(wp + 16);
    }
    if (k0 + 32 < K) __builtin_prefetch(aBase + k0 + 32, 0, 1);

    const v16bf af = __builtin_shufflevector(a_lo, a_hi,
        0,1,2,3,4,5,6,7,8,9,10,11,12,13,14,15);
#pragma unroll
    for (int j = 0; j < NT; ++j) {
      const v16bf bfr = __builtin_shufflevector(blo[j], bhi[j],
          0,1,2,3,4,5,6,7,8,9,10,11,12,13,14,15);
      acc[j] = __builtin_amdgcn_wmma_f32_16x16x32_bf16(
          false, af, false, bfr, (short)0, acc[j], false, false);
    }
  }

  const int crow0 = m0 + ((lane < 16) ? 0 : 8);
#pragma unroll
  for (int j = 0; j < NT; ++j) {
    const int n = n0 + j * 16 + bcol;
    const float bmod = (ACT == 1) ? bias[n] : 0.0f;
#pragma unroll
    for (int r = 0; r < 8; ++r) {
      float v = acc[j][r];
      if (ACT == 1) v = softplus_f(v + bmod);
      C[(size_t)(crow0 + r) * ldc + n] = v;
      if (Cb) Cb[(size_t)(crow0 + r) * ldc + n] = (__bf16)v;
    }
  }
}

// ---------------- depthwise causal conv(4) + SiLU; writes f32 + bf16 -----------------
__global__ __launch_bounds__(256)
void conv_silu_kernel(const float* __restrict__ xz, const float* __restrict__ conv_w,
                      const float* __restrict__ conv_b, float* __restrict__ u,
                      __bf16* __restrict__ ub) {
  const int idx = blockIdx.x * 256 + threadIdx.x;
  const int t = idx >> 11;
  const int d = idx & (D_INNER - 1);
  float acc = conv_b[d];
#pragma unroll
  for (int j = 0; j < D_CONV; ++j) {
    const int tt = t - (D_CONV - 1) + j;
    if (tt >= 0) acc += conv_w[d * D_CONV + j] * xz[(size_t)tt * (2 * D_INNER) + d];
  }
  const float v = silu_f(acc);
  u[idx]  = v;
  ub[idx] = (__bf16)v;
}

// ---------------- selective scan: lane-per-channel, fused gating, bf16 out -----------
__global__ __launch_bounds__(128)
void scan_gate_kernel(const float* __restrict__ xdbl, const float* __restrict__ delta,
                      const float* __restrict__ u, const float* __restrict__ xz,
                      const float* __restrict__ A_log, const float* __restrict__ Dv,
                      __bf16* __restrict__ yg) {
  const int d = blockIdx.x * 128 + threadIdx.x;
  float Ad[D_STATE];
#pragma unroll
  for (int n = 0; n < D_STATE; ++n) Ad[n] = -expf(A_log[d * D_STATE + n]);
  float h[D_STATE];
#pragma unroll
  for (int n = 0; n < D_STATE; ++n) h[n] = 0.0f;
  const float Dd = Dv[d];

  for (int t = 0; t < SEQLEN; ++t) {
    const float* row = xdbl + (size_t)t * (DT_RANK + 2 * D_STATE);
    const float4 B0 = *(const float4*)(row + DT_RANK +  0);
    const float4 B1 = *(const float4*)(row + DT_RANK +  4);
    const float4 B2 = *(const float4*)(row + DT_RANK +  8);
    const float4 B3 = *(const float4*)(row + DT_RANK + 12);
    const float4 C0 = *(const float4*)(row + DT_RANK + 16);
    const float4 C1 = *(const float4*)(row + DT_RANK + 20);
    const float4 C2 = *(const float4*)(row + DT_RANK + 24);
    const float4 C3 = *(const float4*)(row + DT_RANK + 28);
    const float Bv[D_STATE] = {B0.x,B0.y,B0.z,B0.w, B1.x,B1.y,B1.z,B1.w,
                               B2.x,B2.y,B2.z,B2.w, B3.x,B3.y,B3.z,B3.w};
    const float Cv[D_STATE] = {C0.x,C0.y,C0.z,C0.w, C1.x,C1.y,C1.z,C1.w,
                               C2.x,C2.y,C2.z,C2.w, C3.x,C3.y,C3.z,C3.w};
    const float dt = delta[(size_t)t * D_INNER + d];
    const float ut = u[(size_t)t * D_INNER + d];
    const float du = dt * ut;
    float y = 0.0f;
#pragma unroll
    for (int n = 0; n < D_STATE; ++n) {
      h[n] = expf(dt * Ad[n]) * h[n] + du * Bv[n];
      y += h[n] * Cv[n];
    }
    const float zt = xz[(size_t)t * (2 * D_INNER) + D_INNER + d];
    yg[(size_t)t * D_INNER + d] = (__bf16)((y + ut * Dd) * silu_f(zt));
  }
}

extern "C" void kernel_launch(void* const* d_in, const int* in_sizes, int n_in,
                              void* d_out, int out_size, void* d_ws, size_t ws_size,
                              hipStream_t stream) {
  const float* x          = (const float*)d_in[0];
  const float* in_proj_w  = (const float*)d_in[1];
  const float* conv_w     = (const float*)d_in[2];
  const float* conv_b     = (const float*)d_in[3];
  const float* x_proj_w   = (const float*)d_in[4];
  const float* dt_proj_w  = (const float*)d_in[5];
  const float* dt_proj_b  = (const float*)d_in[6];
  const float* A_log      = (const float*)d_in[7];
  const float* Dv         = (const float*)d_in[8];
  const float* out_proj_w = (const float*)d_in[9];
  float* out = (float*)d_out;

  char* p = (char*)d_ws;
  float* xz    = (float*)p; p += (size_t)SEQLEN * 2 * D_INNER * 4;
  float* u     = (float*)p; p += (size_t)SEQLEN * D_INNER * 4;
  float* xdbl  = (float*)p; p += (size_t)SEQLEN * (DT_RANK + 2*D_STATE) * 4;
  float* delta = (float*)p; p += (size_t)SEQLEN * D_INNER * 4;
  __bf16* xb      = (__bf16*)p; p += (size_t)SEQLEN * D_MODEL * 2;
  __bf16* wb_in   = (__bf16*)p; p += (size_t)2 * D_INNER * D_MODEL * 2;
  __bf16* wb_x    = (__bf16*)p; p += (size_t)(DT_RANK + 2*D_STATE) * D_INNER * 2;
  __bf16* wb_dt   = (__bf16*)p; p += (size_t)D_INNER * DT_RANK * 2;
  __bf16* wb_out  = (__bf16*)p; p += (size_t)D_MODEL * D_INNER * 2;
  __bf16* ub      = (__bf16*)p; p += (size_t)SEQLEN * D_INNER * 2;
  __bf16* xdbl_b  = (__bf16*)p; p += (size_t)SEQLEN * (DT_RANK + 2*D_STATE) * 2;
  __bf16* yg_b    = (__bf16*)p; p += (size_t)SEQLEN * D_INNER * 2;

  // 0) one-time bf16 casts
  cast_f32_to_bf16<<<(SEQLEN * D_MODEL) / 2048, 256, 0, stream>>>(x, xb);
  cast_f32_to_bf16<<<(2 * D_INNER * D_MODEL) / 2048, 256, 0, stream>>>(in_proj_w, wb_in);
  cast_f32_to_bf16<<<((DT_RANK + 2*D_STATE) * D_INNER) / 2048, 256, 0, stream>>>(x_proj_w, wb_x);
  cast_f32_to_bf16<<<(D_INNER * DT_RANK) / 2048, 256, 0, stream>>>(dt_proj_w, wb_dt);
  cast_f32_to_bf16<<<(D_MODEL * D_INNER) / 2048, 256, 0, stream>>>(out_proj_w, wb_out);

  // 1) xz = x @ in_proj_w^T   M=2048 N=4096 K=1024  (LDS-staged, 128x64 blocks)
  gemm_bf16_wmma_lds<0><<<(SEQLEN / 128) * (2 * D_INNER / 64), 256, 0, stream>>>(
      xb, wb_in, xz, nullptr, SEQLEN, 2 * D_INNER, D_MODEL,
      D_MODEL, D_MODEL, 2 * D_INNER, nullptr);

  // 2) depthwise conv + SiLU -> u (f32 + bf16)
  conv_silu_kernel<<<(SEQLEN * D_INNER) / 256, 256, 0, stream>>>(xz, conv_w, conv_b, u, ub);

  // 3) x_dbl = u @ x_proj_w^T   M=2048 N=96 K=2048  (per-wave kernel, NT=2)
  gemm_bf16_wmma<2, 0><<<((SEQLEN / 16) * (96 / 32)) / 8, 256, 0, stream>>>(
      ub, wb_x, xdbl, xdbl_b, SEQLEN, DT_RANK + 2 * D_STATE, D_INNER,
      D_INNER, D_INNER, DT_RANK + 2 * D_STATE, nullptr);

  // 4) delta = softplus(x_dbl[:, :64] @ dt_proj_w^T + b)   M=2048 N=2048 K=64
  gemm_bf16_wmma_lds<1><<<(SEQLEN / 128) * (D_INNER / 64), 256, 0, stream>>>(
      xdbl_b, wb_dt, delta, nullptr, SEQLEN, D_INNER, DT_RANK,
      DT_RANK + 2 * D_STATE, DT_RANK, D_INNER, dt_proj_b);

  // 5) selective scan + gating -> yg (bf16)
  scan_gate_kernel<<<D_INNER / 128, 128, 0, stream>>>(xdbl, delta, u, xz, A_log, Dv, yg_b);

  // 6) out = yg @ out_proj_w^T   M=2048 N=1024 K=2048
  gemm_bf16_wmma_lds<0><<<(SEQLEN / 128) * (D_MODEL / 64), 256, 0, stream>>>(
      yg_b, wb_out, out, nullptr, SEQLEN, D_MODEL, D_INNER,
      D_INNER, D_INNER, D_MODEL, nullptr);
}
